// TSDecoder_29145648071100
// MI455X (gfx1250) — compile-verified, matrix operand order checked by reading, not run on hardware
//
#include <hip/hip_runtime.h>
#include <hip/hip_bf16.h>

// ---------------------------------------------------------------------------
// Problem: out[i,j] = sigmoid(W*(||xi||^2+||xj||^2-2*xi.xj)+b), diag zeroed.
// N=8192, D=512.  Strategy:
//   pass 1: fp32 -> bf16 convert (RNE) into workspace + row squared norms
//   pass 2: X*X^T via v_wmma_f32_16x16x32_bf16, fused epilogue
// Workspace layout: [0, 8MB) = X in bf16 (8192*512*2B); [8MB, +32KB) = sq[N].
// ---------------------------------------------------------------------------

typedef __attribute__((ext_vector_type(16))) __bf16          v16bf;
typedef __attribute__((ext_vector_type(8)))  float           v8f;
typedef __attribute__((ext_vector_type(8)))  unsigned short  v8us;

#define TS_N 8192
#define TS_D 512

// round-to-nearest-even fp32 -> bf16 (inputs are finite random normals)
__device__ __forceinline__ unsigned short f32_to_bf16(float f) {
    unsigned int u = __float_as_uint(f);
    u += 0x7FFFu + ((u >> 16) & 1u);
    return (unsigned short)(u >> 16);
}

// ---------------------------------------------------------------------------
// Pass 1: one 256-thread block per row. Each thread converts 2 elements and
// contributes to the squared-norm reduction (wave32 shfl + LDS across 8 waves).
// ---------------------------------------------------------------------------
__global__ void ts_prep_kernel(const float* __restrict__ X,
                               unsigned short* __restrict__ Xb,
                               float* __restrict__ sq) {
    const int row = blockIdx.x;
    const int t   = threadIdx.x;                 // 0..255
    const float2 v = ((const float2*)(X + (size_t)row * TS_D))[t];

    ushort2 o;
    o.x = f32_to_bf16(v.x);
    o.y = f32_to_bf16(v.y);
    ((ushort2*)(Xb + (size_t)row * TS_D))[t] = o;

    float s = v.x * v.x + v.y * v.y;
    #pragma unroll
    for (int off = 16; off > 0; off >>= 1)
        s += __shfl_xor(s, off, 32);             // wave32 reduction

    __shared__ float red[8];
    if ((t & 31) == 0) red[t >> 5] = s;
    __syncthreads();
    if (t == 0) {
        float tot = 0.f;
        #pragma unroll
        for (int i = 0; i < 8; ++i) tot += red[i];
        sq[row] = tot;
    }
}

// ---------------------------------------------------------------------------
// Load one 16x32 bf16 WMMA fragment for memory rows [rowBase, rowBase+16),
// K-slab [k0, k0+32). ISA layout (CDNA5 ISA, 16-bit A 16x32):
//   lane L: row = L%16; half = L/16
//   vector elems 0..7  <- K = k0 + half*8 + (0..7)
//   vector elems 8..15 <- K = k0 + 16 + half*8 + (0..7)
// Two 16B row-major loads per lane; B fragments (gram -> B rows are also rows
// of X) use the identical pattern with row = output column.
// ---------------------------------------------------------------------------
__device__ __forceinline__ v16bf load_frag(const unsigned short* __restrict__ Xb,
                                           int rowBase, int k0,
                                           int lane16, int half) {
    const unsigned short* p = Xb + (size_t)(rowBase + lane16) * TS_D + k0 + half * 8;
    union { v16bf bf; v8us h[2]; } u;
    u.h[0] = *(const v8us*)(p);
    u.h[1] = *(const v8us*)(p + 16);
    return u.bf;
}

// ---------------------------------------------------------------------------
// Pass 2: 64x64 grid of 256-thread blocks (8 waves). Block tile = 128x128.
// Wave grid 2x4: wave computes 64 rows x 32 cols = 4x2 WMMA tiles.
// ---------------------------------------------------------------------------
__global__ void __launch_bounds__(256)
ts_gram_sigmoid_kernel(const unsigned short* __restrict__ Xb,
                       const float* __restrict__ sq,
                       const float* __restrict__ Wp,
                       const float* __restrict__ bp,
                       float* __restrict__ out) {
    const int lane   = threadIdx.x & 31;
    const int wid    = threadIdx.x >> 5;         // 0..7
    const int lane16 = lane & 15;
    const int half   = lane >> 4;

    const int waveRow = wid >> 2;                // 0..1  (64 rows each)
    const int waveCol = wid & 3;                 // 0..3  (32 cols each)
    const int rowBase = blockIdx.y * 128 + waveRow * 64;
    const int colBase = blockIdx.x * 128 + waveCol * 32;

    const float w  = Wp[0];
    const float bb = bp[0];

    const v8f vzero = {0.f, 0.f, 0.f, 0.f, 0.f, 0.f, 0.f, 0.f};
    v8f acc[4][2];
    #pragma unroll
    for (int i = 0; i < 4; ++i)
        #pragma unroll
        for (int j = 0; j < 2; ++j) acc[i][j] = vzero;

    for (int k0 = 0; k0 < TS_D; k0 += 32) {
        // prefetch next K-slab (global_prefetch_b8; data is L2-resident,
        // this pulls it toward the WGP)
        if (k0 + 32 < TS_D) {
            __builtin_prefetch(Xb + (size_t)(rowBase + lane16) * TS_D + k0 + 32, 0, 1);
            __builtin_prefetch(Xb + (size_t)(colBase + lane16) * TS_D + k0 + 32, 0, 1);
        }

        v16bf afrag[4];
        #pragma unroll
        for (int i = 0; i < 4; ++i)
            afrag[i] = load_frag(Xb, rowBase + i * 16, k0, lane16, half);

        v16bf bfrag[2];
        #pragma unroll
        for (int j = 0; j < 2; ++j)
            bfrag[j] = load_frag(Xb, colBase + j * 16, k0, lane16, half);

        #pragma unroll
        for (int i = 0; i < 4; ++i)
            #pragma unroll
            for (int j = 0; j < 2; ++j)
                acc[i][j] = __builtin_amdgcn_wmma_f32_16x16x32_bf16(
                    /*neg_a=*/false, afrag[i],
                    /*neg_b=*/false, bfrag[j],
                    /*c_mod=*/(short)0, acc[i][j],
                    /*reuse_a=*/false, /*reuse_b=*/false);
    }

    // Epilogue. C/D layout: elem r of v8f in lane L -> row = r + 8*half,
    // col = L%16 (within the 16x16 tile).
    #pragma unroll
    for (int j = 0; j < 2; ++j) {
        const int col   = colBase + j * 16 + lane16;
        const float sqc = sq[col];
        #pragma unroll
        for (int i = 0; i < 4; ++i) {
            #pragma unroll
            for (int r = 0; r < 8; ++r) {
                const int row = rowBase + i * 16 + r + 8 * half;
                const float dist2 = sq[row] + sqc - 2.0f * acc[i][j][r];
                float val = 1.0f / (1.0f + __expf(-(w * dist2 + bb)));
                if (row == col) val = 0.0f;
                out[(size_t)row * TS_N + col] = val;
            }
        }
    }
}

// ---------------------------------------------------------------------------
// d_in order (setup_inputs): [0]=node_embs (8192*512 f32), [1]=graph_embs
// (unused by reference), [2]=W (1 f32), [3]=b (1 f32). Output: 8192*8192 f32.
// ---------------------------------------------------------------------------
extern "C" void kernel_launch(void* const* d_in, const int* in_sizes, int n_in,
                              void* d_out, int out_size, void* d_ws, size_t ws_size,
                              hipStream_t stream) {
    (void)in_sizes; (void)n_in; (void)out_size; (void)ws_size;

    const float* X  = (const float*)d_in[0];
    const float* Wp = (const float*)d_in[2];
    const float* bp = (const float*)d_in[3];
    float* out = (float*)d_out;

    unsigned short* Xb = (unsigned short*)d_ws;                              // 8 MB
    float* sq = (float*)((char*)d_ws + (size_t)TS_N * TS_D * sizeof(unsigned short));

    ts_prep_kernel<<<TS_N, 256, 0, stream>>>(X, Xb, sq);
    ts_gram_sigmoid_kernel<<<dim3(TS_N / 128, TS_N / 128), 256, 0, stream>>>(
        Xb, sq, Wp, bp, out);
}
